// RetentiveNetwork_35665408426495
// MI455X (gfx1250) — matrix operand order repaired and checked by reading, not compiled, not run on hardware
//
#include <hip/hip_runtime.h>
#include <hip/hip_bf16.h>
#include <math.h>

// ---------------------------------------------------------------------------
// Types for CDNA5 WMMA (wave32): v_wmma_f32_16x16x32_bf16
// ---------------------------------------------------------------------------
typedef __attribute__((ext_vector_type(16))) __bf16       v16bf;
typedef __attribute__((ext_vector_type(8)))  float        v8f;
typedef __attribute__((ext_vector_type(4)))  unsigned int u32x4;

union V16U { u32x4 u[2]; v16bf v; };

__device__ __forceinline__ unsigned short f2bf(float f) {
  unsigned int u = __builtin_bit_cast(unsigned int, f);
  u += 0x7FFFu + ((u >> 16) & 1u);          // round-to-nearest-even
  return (unsigned short)(u >> 16);
}

// Low 32 bits of a generic pointer to __shared__ == LDS byte offset
// (ISA 10.2: LDS aperture; the LDS address is the low 32 bits).
__device__ __forceinline__ unsigned lds_addr32(const void* p) {
  return (unsigned)(uintptr_t)p;
}

// GLOBAL_LOAD_ASYNC_TO_LDS_B128 (GV mode, SADDR=NULL): per-lane 16B
// global -> LDS transfer tracked with ASYNCcnt.
__device__ __forceinline__ void async_load_b128(unsigned lds, const void* g) {
  asm volatile("global_load_async_to_lds_b128 %0, %1, off"
               :: "v"(lds), "v"((unsigned long long)(uintptr_t)g)
               : "memory");
}

static constexpr int kB  = 2;
static constexpr int kT  = 2048;
static constexpr int kD  = 1024;
static constexpr int kH  = 8;
static constexpr int kHD = 128;

// ---------------------------------------------------------------------------
// fp32 -> bf16 flat conversion
// ---------------------------------------------------------------------------
__global__ void cvt_f32_to_bf16_kernel(const float* __restrict__ in,
                                       unsigned short* __restrict__ out, int n) {
  int i = blockIdx.x * blockDim.x + threadIdx.x;
  if (i < n) out[i] = f2bf(in[i]);
}

// maskv[d] = (1/H) * sum_h exp(-theta_h * d)
__global__ void decay_mask_kernel(const float* __restrict__ theta,
                                  float* __restrict__ maskv, int t) {
  int d = blockIdx.x * blockDim.x + threadIdx.x;
  if (d >= t) return;
  float s = 0.0f;
#pragma unroll
  for (int h = 0; h < kH; ++h) s += __expf(-theta[h] * (float)d);
  maskv[d] = s * (1.0f / (float)kH);
}

// K (B,T,D) bf16 -> Kt (B,H,HD,T) bf16
__global__ void transpose_heads_kernel(const unsigned short* __restrict__ Kin,
                                       unsigned short* __restrict__ Kt) {
  int i = blockIdx.x * blockDim.x + threadIdx.x;   // over B*T*D
  int d  = i % kD;
  int bt = i / kD;
  int t  = bt % kT;
  int b  = bt / kT;
  int h  = d / kHD, e = d % kHD;
  Kt[(((size_t)b * kH + h) * kHD + e) * kT + t] = Kin[i];
}

// ---------------------------------------------------------------------------
// Tiled bf16 GEMM: C[M,N] = A[M,K] @ B[K,N]  (A,B bf16 row-major, f32 accum)
// Block tile 128x64, 8 waves (4x2), each wave 32x32 = 2x2 WMMA frags.
// Double-buffered LDS staging via GLOBAL_LOAD_ASYNC_TO_LDS_B128:
// tile k+1 streams in while tile k feeds the WMMAs.
// ---------------------------------------------------------------------------
template <bool F32OUT>
__global__ __launch_bounds__(256)
void gemm_bf16_kernel(const unsigned short* __restrict__ A,
                      const unsigned short* __restrict__ Bw,
                      void* __restrict__ C,
                      const float* __restrict__ bias,
                      int M, int N, int K) {
  constexpr int LDA = 40;   // padded row stride (elements), rows 80B = 5x16B
  constexpr int LDB = 72;   // padded row stride (elements), rows 144B = 9x16B
  __shared__ __align__(16) unsigned short sA[2][128 * LDA];
  __shared__ __align__(16) unsigned short sB[2][32 * LDB];

  const int tid  = threadIdx.x;
  const int lane = tid & 31;
  const int w    = tid >> 5;
  const int wm   = w >> 1;         // 0..3
  const int wn   = w & 1;          // 0..1
  const int half = lane >> 4;
  const int lr   = lane & 15;

  const int m0 = blockIdx.y * 128;
  const int n0 = blockIdx.x * 64;

  // Per-thread staging slots: A tile 128x32 = 512 b128 chunks (2/thread),
  // B tile 32x64 = 256 b128 chunks (1/thread).
  const int rA = tid >> 2, cA = tid & 3;       // A rows rA and rA+64
  const int rB = tid >> 3, cB = tid & 7;
  const unsigned short* gA0 = A + (size_t)(m0 + rA) * K + cA * 8;
  const unsigned short* gA1 = A + (size_t)(m0 + rA + 64) * K + cA * 8;
  const unsigned short* gB  = Bw + (size_t)rB * N + n0 + cB * 8;
  unsigned lA0[2], lA1[2], lB[2];
#pragma unroll
  for (int bub = 0; bub < 2; ++bub) {
    lA0[bub] = lds_addr32(&sA[bub][rA * LDA + cA * 8]);
    lA1[bub] = lds_addr32(&sA[bub][(rA + 64) * LDA + cA * 8]);
    lB[bub]  = lds_addr32(&sB[bub][rB * LDB + cB * 8]);
  }

  auto issue_tile = [&](int kt) {
    int bub = kt & 1;
    size_t ka = (size_t)kt * 32;            // A advances 32 columns
    size_t kb = (size_t)kt * 32 * (size_t)N; // B advances 32 rows
    async_load_b128(lA0[bub], gA0 + ka);
    async_load_b128(lA1[bub], gA1 + ka);
    async_load_b128(lB[bub],  gB + kb);
  };

  v8f acc[2][2];
#pragma unroll
  for (int i = 0; i < 2; ++i)
#pragma unroll
    for (int j = 0; j < 2; ++j)
#pragma unroll
      for (int r = 0; r < 8; ++r) acc[i][j][r] = 0.0f;

  const int nk = K / 32;
  issue_tile(0);

  for (int kt = 0; kt < nk; ++kt) {
    if (kt + 1 < nk) {
      issue_tile(kt + 1);                                // prefetch next tile
      asm volatile("s_wait_asynccnt 0x3" ::: "memory");  // current tile resident
    } else {
      asm volatile("s_wait_asynccnt 0x0" ::: "memory");
    }
    __syncthreads();

    const unsigned short* cA_ = sA[kt & 1];
    const unsigned short* cB_ = sB[kt & 1];

    // A frag: lane holds row M=lr; K pairs at half*8.. and 16+half*8..
    V16U a[2], b[2];
#pragma unroll
    for (int i = 0; i < 2; ++i) {
      const unsigned short* p = cA_ + (wm * 32 + i * 16 + lr) * LDA;
      a[i].u[0] = *(const u32x4*)(p + half * 8);
      a[i].u[1] = *(const u32x4*)(p + 16 + half * 8);
    }
    // B frag: lane holds col N=lr; K at half*8.. and 16+half*8..
#pragma unroll
    for (int j = 0; j < 2; ++j) {
      const unsigned short* p = cB_ + (half * 8) * LDB + wn * 32 + j * 16 + lr;
#pragma unroll
      for (int s = 0; s < 8; ++s) {
        b[j].v[s] = __builtin_bit_cast(__bf16, p[s * LDB]);
        b[j].v[s + 8] = __builtin_bit_cast(__bf16, p[(16 + s) * LDB]);
      }
    }
#pragma unroll
    for (int i = 0; i < 2; ++i)
#pragma unroll
      for (int j = 0; j < 2; ++j)
        acc[i][j] = __builtin_amdgcn_wmma_f32_16x16x32_bf16(
            false, a[i].v, false, b[j].v, (short)0, acc[i][j], false, false);
    __syncthreads();
  }

  // C layout: lanes 0-15 rows 0-7 (VGPR r), lanes 16-31 rows 8-15; col = lr
#pragma unroll
  for (int i = 0; i < 2; ++i)
#pragma unroll
    for (int j = 0; j < 2; ++j)
#pragma unroll
      for (int r = 0; r < 8; ++r) {
        int gm = m0 + wm * 32 + i * 16 + r + half * 8;
        int gn = n0 + wn * 32 + j * 16 + lr;
        if constexpr (F32OUT)
          ((float*)C)[(size_t)gm * N + gn] = acc[i][j][r] + bias[gn];
        else
          ((unsigned short*)C)[(size_t)gm * N + gn] = f2bf(acc[i][j][r]);
      }
}

// ---------------------------------------------------------------------------
// Retention (flash-style) + softmax + GroupNorm.
// Block = 128 threads = 4 independent waves; each wave owns a 16-query tile
// of one (b,h). Loops over 32-key blocks with online softmax.
// ---------------------------------------------------------------------------
__global__ __launch_bounds__(128)
void retention_kernel(const unsigned short* __restrict__ Q,    // (B,T,D) bf16
                      const unsigned short* __restrict__ Kt,   // (B,H,HD,T) bf16
                      const unsigned short* __restrict__ V,    // (B,T,D) bf16
                      const float* __restrict__ maskv,         // (T)
                      const float* __restrict__ gamma,
                      const float* __restrict__ beta,
                      unsigned short* __restrict__ On) {       // (B,T,D) bf16
  __shared__ __align__(16) unsigned short sP[4][16 * 32];      // wave-private P tiles

  const int tid  = threadIdx.x;
  const int w    = tid >> 5;
  const int lane = tid & 31;
  const int half = lane >> 4;
  const int lr   = lane & 15;
  const int b    = blockIdx.z, h = blockIdx.y;
  const int q0   = blockIdx.x * 64 + w * 16;
  const float scale = 0.08838834764831845f;                    // 1/sqrt(128)

  // Q fragments: 16 rows x 128 hd as 4 A-frags (16x32)
  V16U qa[4];
  {
    const unsigned short* qp = Q + (size_t)(b * kT + q0 + lr) * kD + h * kHD;
#pragma unroll
    for (int c = 0; c < 4; ++c) {
      qa[c].u[0] = *(const u32x4*)(qp + c * 32 + half * 8);
      qa[c].u[1] = *(const u32x4*)(qp + c * 32 + 16 + half * 8);
    }
  }

  v8f acc[8];
#pragma unroll
  for (int c = 0; c < 8; ++c)
#pragma unroll
    for (int r = 0; r < 8; ++r) acc[c][r] = 0.0f;
  float mrow[8], lrow[8];
#pragma unroll
  for (int r = 0; r < 8; ++r) { mrow[r] = -__builtin_inff(); lrow[r] = 0.0f; }

  const unsigned short* ktb = Kt + (size_t)(b * kH + h) * kHD * kT;
  const unsigned short* vbb = V + (size_t)b * kT * kD + h * kHD;
  unsigned short* sp = sP[w];

  for (int kb = 0; kb < q0 + 16; kb += 32) {
    float z[2][8];
#pragma unroll
    for (int t = 0; t < 2; ++t) {
      v8f s;
#pragma unroll
      for (int r = 0; r < 8; ++r) s[r] = 0.0f;
#pragma unroll
      for (int c = 0; c < 4; ++c) {
        // K^T B-frag: lane = key col (kb+t*16+lr), hd at half*8.. and 16+half*8..
        V16U kf;
        const unsigned short* kp = ktb + (size_t)(c * 32 + half * 8) * kT + kb + t * 16 + lr;
#pragma unroll
        for (int s = 0; s < 8; ++s) {
          kf.v[s] = __builtin_bit_cast(__bf16, kp[(size_t)s * kT]);
          kf.v[s + 8] = __builtin_bit_cast(__bf16, kp[(size_t)(16 + s) * kT]);
        }
        s = __builtin_amdgcn_wmma_f32_16x16x32_bf16(false, qa[c].v, false, kf.v,
                                                    (short)0, s, false, false);
      }
      int j = kb + t * 16 + lr;
#pragma unroll
      for (int r = 0; r < 8; ++r) {
        int i = q0 + r + half * 8;
        z[t][r] = (j <= i) ? s[r] * scale * maskv[i - j] : -__builtin_inff();
      }
    }
    // online softmax across these 32 columns (16-lane row groups)
#pragma unroll
    for (int r = 0; r < 8; ++r) {
      float mx = fmaxf(z[0][r], z[1][r]);
      for (int m = 1; m < 16; m <<= 1) mx = fmaxf(mx, __shfl_xor(mx, m, 32));
      float mnew = fmaxf(mrow[r], mx);
      float corr = __expf(mrow[r] - mnew);    // first block: exp(-inf)=0
      float p0 = __expf(z[0][r] - mnew);
      float p1 = __expf(z[1][r] - mnew);
      z[0][r] = p0; z[1][r] = p1;
      float rs = p0 + p1;
      for (int m = 1; m < 16; m <<= 1) rs += __shfl_xor(rs, m, 32);
      lrow[r] = lrow[r] * corr + rs;
      mrow[r] = mnew;
#pragma unroll
      for (int c = 0; c < 8; ++c) acc[c][r] *= corr;
    }
    // P (C-layout) -> LDS -> A-fragment layout
#pragma unroll
    for (int t = 0; t < 2; ++t)
#pragma unroll
      for (int r = 0; r < 8; ++r)
        sp[(r + half * 8) * 32 + t * 16 + lr] = f2bf(z[t][r]);
    asm volatile("s_wait_dscnt 0" ::: "memory");
    V16U pa;
    {
      const unsigned short* pp = sp + lr * 32;
      pa.u[0] = *(const u32x4*)(pp + half * 8);
      pa.u[1] = *(const u32x4*)(pp + 16 + half * 8);
    }
    // out += P @ V : B-frag lane = hd col (c*16+lr), keys at half*8.. and 16+half*8..
#pragma unroll
    for (int c = 0; c < 8; ++c) {
      V16U vf;
      const unsigned short* vp = vbb + (size_t)(kb + half * 8) * kD + c * 16 + lr;
#pragma unroll
      for (int s = 0; s < 8; ++s) {
        vf.v[s] = __builtin_bit_cast(__bf16, vp[(size_t)s * kD]);
        vf.v[s + 8] = __builtin_bit_cast(__bf16, vp[(size_t)(16 + s) * kD]);
      }
      acc[c] = __builtin_amdgcn_wmma_f32_16x16x32_bf16(false, pa.v, false, vf.v,
                                                       (short)0, acc[c], false, false);
    }
  }

  // softmax denom + GroupNorm over HD=128 per row
  float s1[8], s2[8];
#pragma unroll
  for (int r = 0; r < 8; ++r) { s1[r] = 0.0f; s2[r] = 0.0f; }
#pragma unroll
  for (int c = 0; c < 8; ++c)
#pragma unroll
    for (int r = 0; r < 8; ++r) {
      float v = acc[c][r] / lrow[r];
      acc[c][r] = v;
      s1[r] += v;
      s2[r] += v * v;
    }
  float mean[8], rstd[8];
#pragma unroll
  for (int r = 0; r < 8; ++r) {
    float a1 = s1[r], a2 = s2[r];
    for (int m = 1; m < 16; m <<= 1) {
      a1 += __shfl_xor(a1, m, 32);
      a2 += __shfl_xor(a2, m, 32);
    }
    float mu  = a1 * (1.0f / kHD);
    float var = a2 * (1.0f / kHD) - mu * mu;
    mean[r] = mu;
    rstd[r] = rsqrtf(var + 1e-5f);
  }
  unsigned short* ob = On + (size_t)b * kT * kD + h * kHD;
#pragma unroll
  for (int c = 0; c < 8; ++c) {
    int col = c * 16 + lr;
    float g  = gamma[h * kHD + col];
    float be = beta[h * kHD + col];
#pragma unroll
    for (int r = 0; r < 8; ++r) {
      int row = q0 + r + half * 8;
      float v = (acc[c][r] - mean[r]) * rstd[r] * g + be;
      ob[(size_t)row * kD + col] = f2bf(v);
    }
  }
}

// ---------------------------------------------------------------------------
// Launch: cvt -> mask -> QKV GEMMs -> K transpose -> retention -> out GEMM
// ---------------------------------------------------------------------------
extern "C" void kernel_launch(void* const* d_in, const int* in_sizes, int n_in,
                              void* d_out, int out_size, void* d_ws, size_t ws_size,
                              hipStream_t stream) {
  (void)in_sizes; (void)n_in; (void)out_size; (void)ws_size;
  const float* x     = (const float*)d_in[0];
  const float* Wq    = (const float*)d_in[1];
  const float* Wk    = (const float*)d_in[2];
  const float* Wv    = (const float*)d_in[3];
  const float* Wo    = (const float*)d_in[4];
  const float* bo    = (const float*)d_in[5];
  const float* theta = (const float*)d_in[6];
  const float* gamma = (const float*)d_in[7];
  const float* beta  = (const float*)d_in[8];
  float* out = (float*)d_out;

  const size_t nXD = (size_t)kB * kT * kD;   // 4M elements
  const size_t nW  = (size_t)kD * kD;        // 1M elements

  char* ws = (char*)d_ws;
  size_t off = 0;
  auto alloc = [&](size_t bytes) -> void* {
    void* p = ws + off;
    off += (bytes + 255) & ~(size_t)255;
    return p;
  };
  unsigned short* xbf = (unsigned short*)alloc(nXD * 2);
  unsigned short* wqb = (unsigned short*)alloc(nW * 2);
  unsigned short* wkb = (unsigned short*)alloc(nW * 2);
  unsigned short* wvb = (unsigned short*)alloc(nW * 2);
  unsigned short* wob = (unsigned short*)alloc(nW * 2);
  unsigned short* Qm  = (unsigned short*)alloc(nXD * 2);
  unsigned short* Km  = (unsigned short*)alloc(nXD * 2);
  unsigned short* Ktm = (unsigned short*)alloc(nXD * 2);
  unsigned short* Vm  = (unsigned short*)alloc(nXD * 2);
  float* maskv        = (float*)alloc((size_t)kT * 4);
  unsigned short* On  = xbf;   // reuse: x dead after QKV GEMMs

  const int thr = 256;
  cvt_f32_to_bf16_kernel<<<(int)(nXD / thr), thr, 0, stream>>>(x, xbf, (int)nXD);
  cvt_f32_to_bf16_kernel<<<(int)(nW / thr), thr, 0, stream>>>(Wq, wqb, (int)nW);
  cvt_f32_to_bf16_kernel<<<(int)(nW / thr), thr, 0, stream>>>(Wk, wkb, (int)nW);
  cvt_f32_to_bf16_kernel<<<(int)(nW / thr), thr, 0, stream>>>(Wv, wvb, (int)nW);
  cvt_f32_to_bf16_kernel<<<(int)(nW / thr), thr, 0, stream>>>(Wo, wob, (int)nW);
  decay_mask_kernel<<<kT / thr, thr, 0, stream>>>(theta, maskv, kT);

  dim3 ggrid(kD / 64, (kB * kT) / 128);
  gemm_bf16_kernel<false><<<ggrid, 256, 0, stream>>>(xbf, wqb, Qm, nullptr,
                                                     kB * kT, kD, kD);
  gemm_bf16_kernel<false><<<ggrid, 256, 0, stream>>>(xbf, wkb, Km, nullptr,
                                                     kB * kT, kD, kD);
  gemm_bf16_kernel<false><<<ggrid, 256, 0, stream>>>(xbf, wvb, Vm, nullptr,
                                                     kB * kT, kD, kD);

  transpose_heads_kernel<<<(int)(nXD / thr), thr, 0, stream>>>(Km, Ktm);

  retention_kernel<<<dim3(kT / 64, kH, kB), 128, 0, stream>>>(
      Qm, Ktm, Vm, maskv, gamma, beta, On);

  gemm_bf16_kernel<true><<<ggrid, 256, 0, stream>>>(On, wob, out, bo,
                                                    kB * kT, kD, kD);
}
